// GraphConvolution_67783173865566
// MI455X (gfx1250) — matrix operand order repaired and checked by reading, not compiled, not run on hardware
//
#include <hip/hip_runtime.h>

// ---------------------------------------------------------------------------
// CDNA5 / gfx1250 types for WMMA
// ---------------------------------------------------------------------------
typedef __attribute__((ext_vector_type(16))) __bf16 v16bf;
typedef __attribute__((ext_vector_type(8)))  float  v8f;

__device__ __forceinline__ unsigned short bf16bits(float f) {
  unsigned u = __builtin_bit_cast(unsigned, f);
  u += 0x7FFFu + ((u >> 16) & 1u);          // round-to-nearest-even
  return (unsigned short)(u >> 16);
}

// ---------------------------------------------------------------------------
// 0) zero scratch (grid-stride float4 stores)
// ---------------------------------------------------------------------------
__global__ void zero_f4(float4* __restrict__ p, long n4) {
  long i = (long)blockIdx.x * blockDim.x + threadIdx.x;
  long stride = (long)gridDim.x * blockDim.x;
  for (; i < n4; i += stride) p[i] = make_float4(0.f, 0.f, 0.f, 0.f);
}

// ---------------------------------------------------------------------------
// 1) pack the 3 weight matrices [256(K) x 256(N)] f32 row-major into the
//    WMMA B-fragment layout (bf16), so GEMM B loads are contiguous 32B/lane.
//    Bpack[((kk*16 + ct)*32 + lane)*16 + e] = bf16(W[k*256 + n])
//    with s = lane>>4, n = ct*16 + (lane&15), k = kk*32 + (e<8 ? s*8+e : 16+s*8+(e-8)).
// ---------------------------------------------------------------------------
__global__ void pack_weights_bf16(const float* __restrict__ wA,
                                  const float* __restrict__ wA2,
                                  const float* __restrict__ wmlp,
                                  unsigned short* __restrict__ Bpack) {
  int t = blockIdx.x * blockDim.x + threadIdx.x;
  if (t >= 3 * 65536) return;
  int m = t >> 16;
  int i = t & 65535;
  int e    = i & 15;
  int lane = (i >> 4) & 31;
  int ct   = (i >> 9) & 15;
  int kk   = i >> 13;
  int s = lane >> 4;
  int n = ct * 16 + (lane & 15);
  int k = kk * 32 + ((e < 8) ? (s * 8 + e) : (16 + s * 8 + (e - 8)));
  const float* W = (m == 0) ? wA : (m == 1) ? wA2 : wmlp;
  Bpack[m * 65536 + i] = bf16bits(W[k * 256 + n]);
}

// ---------------------------------------------------------------------------
// 2) fused triple GEMM: Y[y] = X @ W[y] for y in {A, A2, mlp(relu)}.
//    Block = 256 threads = 8 waves. Tile = 64 rows x 256 cols.
//    Wave (band = w&3, colgroup = w>>2): 16-row band, 8 col-tiles of 16.
//    K loop: 8 steps of K=32 with v_wmma_f32_16x16x32_bf16.
// ---------------------------------------------------------------------------
__global__ __launch_bounds__(256) void gemm3_wmma(
    const float* __restrict__ X, const unsigned short* __restrict__ Bpack,
    float* __restrict__ YA, float* __restrict__ YA2, float* __restrict__ Ymlp,
    int n) {
  const int wsel = blockIdx.y;  // 0: A, 1: A2, 2: mlp
  const unsigned short* Bw = Bpack + wsel * 65536;
  float* Y = (wsel == 0) ? YA : (wsel == 1) ? YA2 : Ymlp;
  const bool do_relu = (wsel == 2);

  const int tid  = threadIdx.x;
  const int wave = tid >> 5;
  const int lane = tid & 31;
  const int band = wave & 3;   // row band within 64-row tile
  const int cg   = wave >> 2;  // 0/1: columns [0,128) / [128,256)
  const int s    = lane >> 4;
  const int lm   = lane & 15;

  const int rowTile = blockIdx.x * 64;
  int arow = rowTile + band * 16 + lm;     // A layout: lane%16 == M
  if (arow >= n) arow = n - 1;             // clamp: keep EXEC all-ones for WMMA
  const float* __restrict__ Arow = X + (size_t)arow * 256;

  v8f acc[8] = {};

  for (int kk = 0; kk < 8; ++kk) {
    // A fragment: elems 0..7 = K kk*32 + s*8 + i ; 8..15 = +16 (ISA 16b A 16x32)
    const float* pa = Arow + kk * 32;
    float4 f0 = *(const float4*)(pa + s * 8);
    float4 f1 = *(const float4*)(pa + s * 8 + 4);
    float4 f2 = *(const float4*)(pa + 16 + s * 8);
    float4 f3 = *(const float4*)(pa + 16 + s * 8 + 4);
    if (kk < 7) __builtin_prefetch(pa + 32, 0, 0);  // global_prefetch next K chunk

    union { v16bf v; unsigned short u[16]; } A;
    A.u[0]  = bf16bits(f0.x); A.u[1]  = bf16bits(f0.y);
    A.u[2]  = bf16bits(f0.z); A.u[3]  = bf16bits(f0.w);
    A.u[4]  = bf16bits(f1.x); A.u[5]  = bf16bits(f1.y);
    A.u[6]  = bf16bits(f1.z); A.u[7]  = bf16bits(f1.w);
    A.u[8]  = bf16bits(f2.x); A.u[9]  = bf16bits(f2.y);
    A.u[10] = bf16bits(f2.z); A.u[11] = bf16bits(f2.w);
    A.u[12] = bf16bits(f3.x); A.u[13] = bf16bits(f3.y);
    A.u[14] = bf16bits(f3.z); A.u[15] = bf16bits(f3.w);

#pragma unroll
    for (int c = 0; c < 8; ++c) {
      int ct = cg * 8 + c;
      const v16bf b = *reinterpret_cast<const v16bf*>(
          Bw + (size_t)((kk * 16 + ct) * 32 + lane) * 16);
      acc[c] = __builtin_amdgcn_wmma_f32_16x16x32_bf16(
          false, A.v, false, b, (short)0, acc[c], false, false);
    }
  }

  // C layout: VGPR j, lane group s -> M = s*8 + j, N = lane&15
#pragma unroll
  for (int c = 0; c < 8; ++c) {
    int col = cg * 128 + c * 16 + lm;
#pragma unroll
    for (int j = 0; j < 8; ++j) {
      int row = rowTile + band * 16 + s * 8 + j;
      if (row < n) {
        float v = acc[c][j];
        if (do_relu) v = fmaxf(v, 0.f);
        Y[(size_t)row * 256 + col] = v;
      }
    }
  }
}

// ---------------------------------------------------------------------------
// 3) COO SpMM via float atomics: acc[r,:] += val * Y[c,:]
//    64 lanes x float4 per edge, 4 edges per 256-thread block.
//    Gather source (102 MB) is L2-resident; atomics resolve in L2.
// ---------------------------------------------------------------------------
__global__ __launch_bounds__(256) void spmm_edge_atomic(
    const int* __restrict__ rowA, const int* __restrict__ colA,
    const float* __restrict__ valA,
    const int* __restrict__ rowB, const int* __restrict__ colB,
    const float* __restrict__ valB,
    const float* __restrict__ YA, const float* __restrict__ YB,
    float* __restrict__ accA, float* __restrict__ accB, int nedges) {
  const int* rows; const int* cols; const float* vals;
  const float* Y; float* acc;
  if (blockIdx.y == 0) { rows = rowA; cols = colA; vals = valA; Y = YA; acc = accA; }
  else                 { rows = rowB; cols = colB; vals = valB; Y = YB; acc = accB; }

  int slot = threadIdx.x >> 6;
  int q    = threadIdx.x & 63;
  long e = (long)blockIdx.x * 4 + slot;
  if (e >= nedges) return;

  int r = rows[e], c = cols[e];
  float v = vals[e];
  const float4 x = *(const float4*)(Y + (size_t)c * 256 + q * 4);
  float* dst = acc + (size_t)r * 256 + q * 4;
  unsafeAtomicAdd(dst + 0, v * x.x);
  unsafeAtomicAdd(dst + 1, v * x.y);
  unsafeAtomicAdd(dst + 2, v * x.z);
  unsafeAtomicAdd(dst + 3, v * x.w);
}

// ---------------------------------------------------------------------------
// 4) in-place relu on the two SpMM accumulators + column sums of all three
//    matrices (for the mean-based attention keys). 128 row slices per matrix.
// ---------------------------------------------------------------------------
__global__ __launch_bounds__(256) void relu_colsum(
    float* __restrict__ accA, float* __restrict__ accB,
    const float* __restrict__ Ymlp, float* __restrict__ colsum, int n) {
  int m     = blockIdx.x >> 7;   // 0..2
  int slice = blockIdx.x & 127;
  int c = threadIdx.x;
  float sum = 0.f;
  if (m == 2) {
    for (int r = slice; r < n; r += 128) sum += Ymlp[(size_t)r * 256 + c];
  } else {
    float* M = (m == 0) ? accA : accB;
    for (int r = slice; r < n; r += 128) {
      size_t idx = (size_t)r * 256 + c;
      float v = fmaxf(M[idx], 0.f);
      M[idx] = v;
      sum += v;
    }
  }
  unsafeAtomicAdd(&colsum[m * 256 + c], sum);
}

// ---------------------------------------------------------------------------
// 5) tiny kernel: k_j = mean_j @ W_kj   (64)
//                 v_j = att_vec_j @ k_j (256)   -> per-node logits are dots
// ---------------------------------------------------------------------------
__global__ __launch_bounds__(256) void attn_keys(
    const float* __restrict__ colsum,
    const float* __restrict__ Wk0, const float* __restrict__ Wk1,
    const float* __restrict__ Wk2,
    const float* __restrict__ avA, const float* __restrict__ avA2,
    const float* __restrict__ avM, float* __restrict__ vvec, int n) {
  __shared__ float kbuf[3][64];
  int t = threadIdx.x;
  if (t < 192) {
    int j = t >> 6, h = t & 63;
    const float* Wk = (j == 0) ? Wk0 : (j == 1) ? Wk1 : Wk2;
    float invn = 1.0f / (float)n;
    float acc = 0.f;
    for (int c = 0; c < 256; ++c)
      acc += colsum[j * 256 + c] * invn * Wk[c * 64 + h];
    kbuf[j][h] = acc;
  }
  __syncthreads();
  int c = t;
  for (int j = 0; j < 3; ++j) {
    const float* av = (j == 0) ? avA : (j == 1) ? avA2 : avM;
    float acc = 0.f;
#pragma unroll
    for (int h = 0; h < 64; ++h) acc += av[c * 64 + h] * kbuf[j][h];
    vvec[j * 256 + c] = acc;
  }
}

// ---------------------------------------------------------------------------
// 6) final gating: one wave32 per node. logit_j = <x_j, v_j> via lane
//    partials + shfl_xor reduction; sigmoid -> @att_vec/3 -> softmax ->
//    out = 3*(a0*x0 + a1*x1 + a2*x2). accA aliases d_out (read row, then
//    overwrite row within the same wave — safe).
// ---------------------------------------------------------------------------
__device__ __forceinline__ float dot8(float4 a, float4 b, float4 va, float4 vb) {
  return a.x * va.x + a.y * va.y + a.z * va.z + a.w * va.w +
         b.x * vb.x + b.y * vb.y + b.z * vb.z + b.w * vb.w;
}

__global__ __launch_bounds__(256) void final_gate(
    float* __restrict__ accA, const float* __restrict__ accB,
    const float* __restrict__ Ymlp, const float* __restrict__ vvec,
    const float* __restrict__ attv, int n) {
  int wave = threadIdx.x >> 5, lane = threadIdx.x & 31;
  int node = blockIdx.x * 8 + wave;
  if (node >= n) return;
  size_t base = (size_t)node * 256 + lane * 8;

  float4 a0 = *(const float4*)(accA + base), a1 = *(const float4*)(accA + base + 4);
  float4 b0 = *(const float4*)(accB + base), b1 = *(const float4*)(accB + base + 4);
  float4 c0 = *(const float4*)(Ymlp + base), c1 = *(const float4*)(Ymlp + base + 4);

  size_t vb = (size_t)lane * 8;
  float d0 = dot8(a0, a1, *(const float4*)(vvec + vb),       *(const float4*)(vvec + vb + 4));
  float d1 = dot8(b0, b1, *(const float4*)(vvec + 256 + vb), *(const float4*)(vvec + 256 + vb + 4));
  float d2 = dot8(c0, c1, *(const float4*)(vvec + 512 + vb), *(const float4*)(vvec + 512 + vb + 4));

#pragma unroll
  for (int off = 16; off > 0; off >>= 1) {
    d0 += __shfl_xor(d0, off, 32);
    d1 += __shfl_xor(d1, off, 32);
    d2 += __shfl_xor(d2, off, 32);
  }

  float g0 = 1.f / (1.f + __expf(-d0));
  float g1 = 1.f / (1.f + __expf(-d1));
  float g2 = 1.f / (1.f + __expf(-d2));
  float s0 = (g0 * attv[0] + g1 * attv[3] + g2 * attv[6]) * (1.f / 3.f);
  float s1 = (g0 * attv[1] + g1 * attv[4] + g2 * attv[7]) * (1.f / 3.f);
  float s2 = (g0 * attv[2] + g1 * attv[5] + g2 * attv[8]) * (1.f / 3.f);
  float mx = fmaxf(s0, fmaxf(s1, s2));
  float e0 = __expf(s0 - mx), e1 = __expf(s1 - mx), e2 = __expf(s2 - mx);
  float inv = 3.f / (e0 + e1 + e2);   // fold the trailing 3.0* into softmax
  float t0 = e0 * inv, t1 = e1 * inv, t2 = e2 * inv;

  float4 o0, o1;
  o0.x = t0 * a0.x + t1 * b0.x + t2 * c0.x;
  o0.y = t0 * a0.y + t1 * b0.y + t2 * c0.y;
  o0.z = t0 * a0.z + t1 * b0.z + t2 * c0.z;
  o0.w = t0 * a0.w + t1 * b0.w + t2 * c0.w;
  o1.x = t0 * a1.x + t1 * b1.x + t2 * c1.x;
  o1.y = t0 * a1.y + t1 * b1.y + t2 * c1.y;
  o1.z = t0 * a1.z + t1 * b1.z + t2 * c1.z;
  o1.w = t0 * a1.w + t1 * b1.w + t2 * c1.w;
  *(float4*)(accA + base)     = o0;
  *(float4*)(accA + base + 4) = o1;
}

// ---------------------------------------------------------------------------
// launch
// ---------------------------------------------------------------------------
extern "C" void kernel_launch(void* const* d_in, const int* in_sizes, int n_in,
                              void* d_out, int out_size, void* d_ws, size_t ws_size,
                              hipStream_t stream) {
  const float* X     = (const float*)d_in[0];
  const int*   rowA  = (const int*)d_in[1];
  const int*   colA  = (const int*)d_in[2];
  const float* valA  = (const float*)d_in[3];
  const int*   rowA2 = (const int*)d_in[4];
  const int*   colA2 = (const int*)d_in[5];
  const float* valA2 = (const float*)d_in[6];
  const float* wA    = (const float*)d_in[7];
  const float* wA2   = (const float*)d_in[8];
  const float* wmlp  = (const float*)d_in[9];
  const float* Wk0   = (const float*)d_in[10];
  const float* Wk1   = (const float*)d_in[11];
  const float* Wk2   = (const float*)d_in[12];
  const float* avA   = (const float*)d_in[13];
  const float* avA2  = (const float*)d_in[14];
  const float* avM   = (const float*)d_in[15];
  const float* attv  = (const float*)d_in[16];

  const int N = in_sizes[0] / 256;
  const int E = in_sizes[1];
  float* out = (float*)d_out;            // doubles as accA

  const size_t NF = (size_t)N * 256;
  float* YA   = (float*)d_ws;
  float* YA2  = YA + NF;
  float* Ymlp = YA2 + NF;
  float* accA2 = Ymlp + NF;
  unsigned short* Bpack = (unsigned short*)(accA2 + NF);  // 3 * 65536 bf16
  float* colsum = (float*)(Bpack + 3 * 65536);            // 768
  float* vvec   = colsum + 768;                           // 768

  // 0) zero accumulators + colsum
  zero_f4<<<2048, 256, 0, stream>>>((float4*)out, (long)(NF / 4));
  zero_f4<<<2048, 256, 0, stream>>>((float4*)accA2, (long)(NF / 4));
  zero_f4<<<1, 192, 0, stream>>>((float4*)colsum, 192);

  // 1) pack weights to bf16 WMMA-B layout
  pack_weights_bf16<<<(3 * 65536 + 255) / 256, 256, 0, stream>>>(wA, wA2, wmlp, Bpack);

  // 2) fused triple WMMA GEMM
  gemm3_wmma<<<dim3((N + 63) / 64, 3), 256, 0, stream>>>(X, Bpack, YA, YA2, Ymlp, N);

  // 3) both SpMMs (edge-parallel atomics)
  spmm_edge_atomic<<<dim3((unsigned)((E + 3) / 4), 2), 256, 0, stream>>>(
      rowA, colA, valA, rowA2, colA2, valA2, YA, YA2, out, accA2, E);

  // 4) relu + column sums
  relu_colsum<<<3 * 128, 256, 0, stream>>>(out, accA2, Ymlp, colsum, N);

  // 5) attention key/value vectors
  attn_keys<<<1, 256, 0, stream>>>(colsum, Wk0, Wk1, Wk2, avA, avA2, avM, vvec, N);

  // 6) gating + weighted sum into d_out
  final_gate<<<(N + 7) / 8, 256, 0, stream>>>(out, accA2, Ymlp, vvec, attv, N);
}
